// Qwen3Attention_7662221656215
// MI455X (gfx1250) — matrix operand order repaired and checked by reading, not compiled, not run on hardware
//
#include <hip/hip_runtime.h>

// ---------------------------------------------------------------------------
// Qwen3 attention layer for MI455X (gfx1250, wave32, WMMA + async LDS DMA).
// Pipeline: f32->bf16 convert (+weight transpose) once, then:
//   bf16 WMMA GEMM (qkv proj) -> rmsnorm+rope -> flash attention (WMMA)
//   -> bf16 WMMA GEMM (output proj).
// All LDS tile fills are GLOBAL_LOAD_ASYNC_TO_LDS_B128 (ASYNCcnt) when the
// toolchain exposes the builtin; GEMM is double-buffered on asynccnt.
// ---------------------------------------------------------------------------

typedef __attribute__((ext_vector_type(16))) __bf16 v16bf;
typedef __attribute__((ext_vector_type(8)))  __bf16 v8bf;
typedef __attribute__((ext_vector_type(4)))  __bf16 v4bf;
typedef __attribute__((ext_vector_type(8)))  float  v8f;
// GCC-style vector to match the async builtin's "V4i" parameter type exactly.
typedef int v4i_g __attribute__((vector_size(16)));

#define WMMA_BF16(a, b, c) \
  __builtin_amdgcn_wmma_f32_16x16x32_bf16(false, (a), false, (b), (short)0, (c), false, false)

static __device__ __forceinline__ v16bf cat8(v8bf lo, v8bf hi) {
  return __builtin_shufflevector(lo, hi, 0,1,2,3,4,5,6,7,8,9,10,11,12,13,14,15);
}

#ifndef __has_builtin
#define __has_builtin(x) 0
#endif
#if __has_builtin(__builtin_amdgcn_global_load_async_to_lds_b128)
#define HAS_ASYNC 1
#else
#define HAS_ASYNC 0
#endif

// 16-byte global -> LDS copy (async DMA when available).
static __device__ __forceinline__ void async_copy16(void* lds_dst, const void* gsrc) {
#if HAS_ASYNC
  __builtin_amdgcn_global_load_async_to_lds_b128(
      (__attribute__((address_space(1))) v4i_g*)(void*)gsrc,
      (__attribute__((address_space(3))) v4i_g*)lds_dst, 0, 0);
#else
  *(uint4*)lds_dst = *(const uint4*)gsrc;
#endif
}

// Wait until at most the newest in-flight tile (8 async ops/wave) is pending.
static __device__ __forceinline__ void wait_async_tile_pending() {
#if HAS_ASYNC
#if __has_builtin(__builtin_amdgcn_s_wait_asynccnt)
  __builtin_amdgcn_s_wait_asynccnt(8);
#else
  asm volatile("s_wait_asynccnt 0x8" ::: "memory");
#endif
#endif
}
static __device__ __forceinline__ void wait_async_all() {
#if HAS_ASYNC
#if __has_builtin(__builtin_amdgcn_s_wait_asynccnt)
  __builtin_amdgcn_s_wait_asynccnt(0);
#else
  asm volatile("s_wait_asynccnt 0x0" ::: "memory");
#endif
#endif
}

constexpr int Bc  = 2;
constexpr int Sc  = 2048;
constexpr int Dc  = 4096;
constexpr int Hc  = 32;
constexpr int KVc = 8;
constexpr int HDc = 128;
constexpr int Gc  = Hc / KVc;

// ---------------------------------------------------------------------------
// One-time converters (bandwidth-bound; tiny vs. the GEMM FLOPs).
// ---------------------------------------------------------------------------
__global__ __launch_bounds__(256) void cvt_bf16_kernel(
    const float* __restrict__ in, __bf16* __restrict__ out, int n4) {
  const int i = blockIdx.x * 256 + threadIdx.x;
  if (i < n4) {
    const float4 f = ((const float4*)in)[i];
    v4bf h; h[0] = (__bf16)f.x; h[1] = (__bf16)f.y; h[2] = (__bf16)f.z; h[3] = (__bf16)f.w;
    ((v4bf*)out)[i] = h;
  }
}

// in: f32 [R][C]  ->  out: bf16 [C][R]   (K-contiguous weights for the GEMM)
__global__ __launch_bounds__(256) void cvt_transpose_bf16_kernel(
    const float* __restrict__ in, __bf16* __restrict__ out, int R, int C) {
  __shared__ __bf16 tile[64][72];
  const int r0 = blockIdx.y * 64, c0 = blockIdx.x * 64;
  for (int t = 0; t < 16; ++t) {
    const int idx = threadIdx.x + t * 256;
    const int r = idx >> 6, c = idx & 63;
    tile[r][c] = (__bf16)in[(size_t)(r0 + r) * C + (c0 + c)];
  }
  __syncthreads();
  for (int t = 0; t < 16; ++t) {
    const int idx = threadIdx.x + t * 256;
    const int c = idx >> 6, r = idx & 63;
    out[(size_t)(c0 + c) * R + (r0 + r)] = tile[r][c];
  }
}

// ---------------------------------------------------------------------------
// C(f32) = A(bf16 [M][K]) * Bt(bf16 [N][K], K-contiguous).
// Block tile 128x128, K tile 64, double-buffered async LDS fills.
// 8 waves in 4x2 grid; each wave 32x64 output = 16 WMMA per K tile.
// ---------------------------------------------------------------------------
__global__ __launch_bounds__(256) void gemm_bf16_wmma(
    const __bf16* __restrict__ A, const __bf16* __restrict__ Bt,
    float* __restrict__ C, int M, int N, int K) {
  constexpr int BM = 128, BN = 128, BK = 64;
  __shared__ __attribute__((aligned(128))) __bf16 sA[2][BM][BK];  // [m][k]
  __shared__ __attribute__((aligned(128))) __bf16 sB[2][BN][BK];  // [n][k]

  const int tid  = threadIdx.x;
  const int wv   = tid >> 5;
  const int lane = tid & 31;
  const int hw   = lane >> 4;
  const int l16  = lane & 15;
  const int bm = blockIdx.y * BM;
  const int bn = blockIdx.x * BN;
  const int wm = (wv & 3) * 32;
  const int wn = (wv >> 2) * 64;

  // Fill one K tile: 1024 16B chunks for A + 1024 for B -> 8 async ops/wave.
  auto fill = [&](int k0, int buf) {
    for (int t = 0; t < 4; ++t) {
      const int idx = tid + t * 256;
      const int r = idx >> 3, c = (idx & 7) * 8;
      async_copy16(&sA[buf][r][c], A + (size_t)(bm + r) * K + k0 + c);
    }
    for (int t = 0; t < 4; ++t) {
      const int idx = tid + t * 256;
      const int r = idx >> 3, c = (idx & 7) * 8;
      async_copy16(&sB[buf][r][c], Bt + (size_t)(bn + r) * K + k0 + c);
    }
  };

  v8f vz = {};
  v8f acc[2][4];
  for (int i = 0; i < 2; i++)
    for (int j = 0; j < 4; j++) acc[i][j] = vz;

  fill(0, 0);
  const int nt = K / BK;
  for (int it = 0; it < nt; ++it) {
    const int buf = it & 1;
    if (it + 1 < nt) {
      fill((it + 1) * BK, buf ^ 1);
      wait_async_tile_pending();     // tile `it` done; tile `it+1` in flight
    } else {
      wait_async_all();
    }
    __syncthreads();

    for (int kk = 0; kk < BK; kk += 32) {
      v16bf af[2], bfm[4];
      for (int t = 0; t < 2; t++) {
        const int row = wm + t * 16 + l16;
        af[t] = *(const v16bf*)&sA[buf][row][kk + hw * 16];
      }
      for (int t = 0; t < 4; t++) {
        const int col = wn + t * 16 + l16;
        bfm[t] = *(const v16bf*)&sB[buf][col][kk + hw * 16];
      }
      for (int i = 0; i < 2; i++)
        for (int j = 0; j < 4; j++)
          acc[i][j] = WMMA_BF16(af[i], bfm[j], acc[i][j]);
    }
    __syncthreads();
  }

  for (int i = 0; i < 2; i++)
    for (int j = 0; j < 4; j++) {
      float* cp = C + (size_t)(bm + wm + i * 16 + 8 * hw) * N + (bn + wn + j * 16 + l16);
      for (int r = 0; r < 8; r++) cp[(size_t)r * N] = acc[i][j][r];
    }
}

// ---------------------------------------------------------------------------
// Per-head RMSNorm + NeoX RoPE. Outputs bf16:
//   Q[b][h][s][d], K[b][kh][s][d], V[b][kh][d][s]  (V pre-transposed!)
// grid = (B*S, H+2*KV), block = 128.
// ---------------------------------------------------------------------------
__global__ __launch_bounds__(128) void norm_rope_kernel(
    const float* __restrict__ qkv, const int* __restrict__ positions,
    const float* __restrict__ qw, const float* __restrict__ kw,
    __bf16* __restrict__ Q, __bf16* __restrict__ K, __bf16* __restrict__ V) {
  const int token = blockIdx.x;
  const int slot  = blockIdx.y;
  const int t     = threadIdx.x;
  const int b = token / Sc;
  const int s = token - b * Sc;

  __shared__ float red[4];
  __shared__ float xs[HDc];

  const float val = qkv[(size_t)token * ((Hc + 2 * KVc) * HDc) + (size_t)slot * HDc + t];

  if (slot >= Hc + KVc) {  // V: transposed [d][s] layout for async attention fills
    const int kh = slot - (Hc + KVc);
    V[(((size_t)b * KVc + kh) * HDc + t) * Sc + s] = (__bf16)val;
    return;
  }

  float ssum = val * val;
  ssum += __shfl_xor(ssum, 1, 32);  ssum += __shfl_xor(ssum, 2, 32);
  ssum += __shfl_xor(ssum, 4, 32);  ssum += __shfl_xor(ssum, 8, 32);
  ssum += __shfl_xor(ssum, 16, 32);
  if ((t & 31) == 0) red[t >> 5] = ssum;
  __syncthreads();
  const float mean = (red[0] + red[1] + red[2] + red[3]) * (1.0f / HDc);
  const float w = (slot < Hc) ? qw[t] : kw[t];
  const float xn = val * rsqrtf(mean + 1e-6f) * w;
  xs[t] = xn;
  __syncthreads();

  const float pos = (float)positions[token];
  const int   j   = t & 63;
  const float inv = __powf(10000.0f, -(float)(2 * j) * (1.0f / HDc));
  const float ang = pos * inv;
  const float cv = __cosf(ang), sv = __sinf(ang);
  const float partner = xs[t ^ 64];
  const float out = (t < 64) ? (xn * cv - partner * sv) : (xn * cv + partner * sv);

  if (slot < Hc) {
    Q[(((size_t)b * Hc + slot) * Sc + s) * HDc + t] = (__bf16)out;
  } else {
    const int kh = slot - Hc;
    K[(((size_t)b * KVc + kh) * Sc + s) * HDc + t] = (__bf16)out;
  }
}

// ---------------------------------------------------------------------------
// Causal GQA flash attention. 128 q-rows/block, 8 waves x 16 complete rows,
// 64-key tiles, online softmax. QK^T and P*V via WMMA; K/V tiles filled with
// async LDS DMA (V already [d][s] so no in-kernel transpose). Output bf16.
// ---------------------------------------------------------------------------
__global__ __launch_bounds__(256) void attn_kernel(
    const __bf16* __restrict__ Qg, const __bf16* __restrict__ Kg,
    const __bf16* __restrict__ Vg, __bf16* __restrict__ Og) {
  constexpr int QB = 128, KB = 64;
  const int b   = blockIdx.z;
  const int h   = blockIdx.y;
  const int kvh = h / Gc;
  const int qb  = blockIdx.x * QB;

  __shared__ __attribute__((aligned(128))) __bf16 sQ[QB][HDc];    // 32 KB
  __shared__ __attribute__((aligned(128))) __bf16 sK[KB][HDc];    // 16 KB [key][d]
  __shared__ __attribute__((aligned(128))) __bf16 sVt[HDc][KB];   // 16 KB [d][key]
  __shared__ __attribute__((aligned(128))) __bf16 sP[8][16][KB];  // 16 KB

  const int tid  = threadIdx.x;
  const int wv   = tid >> 5;
  const int lane = tid & 31;
  const int hw   = lane >> 4;
  const int l16  = lane & 15;

  const __bf16* Qbase = Qg + (((size_t)b * Hc + h) * Sc + qb) * HDc;
  const __bf16* Kbase = Kg + (((size_t)b * KVc + kvh) * Sc) * HDc;
  const __bf16* Vbase = Vg + (((size_t)b * KVc + kvh) * HDc) * Sc;

  // Q tile: 2048 16B chunks (8/thread), async
  for (int t = 0; t < 8; ++t) {
    const int idx = tid + t * 256;
    const int r = idx >> 4, c = (idx & 15) * 8;
    async_copy16(&sQ[r][c], Qbase + (size_t)r * HDc + c);
  }

  v8f vz = {};
  float m_run[8], l_run[8];
  v8f oacc[8];
  for (int i = 0; i < 8; i++) { m_run[i] = -3.0e38f; l_run[i] = 0.0f; oacc[i] = vz; }

  const int   rowa  = qb + wv * 16 + 8 * hw;
  const float scale = 0.08838834764831845f;  // 1/sqrt(128)

  for (int kt = 0; kt < qb + QB; kt += KB) {
    // K tile: 1024 chunks (4/thread)
    for (int t = 0; t < 4; ++t) {
      const int idx = tid + t * 256;
      const int r = idx >> 4, c = (idx & 15) * 8;
      async_copy16(&sK[r][c], Kbase + (size_t)(kt + r) * HDc + c);
    }
    // V^T tile: rows d, key segment [kt, kt+64): 1024 chunks (4/thread)
    for (int t = 0; t < 4; ++t) {
      const int idx = tid + t * 256;
      const int r = idx >> 3, c = (idx & 7) * 8;
      async_copy16(&sVt[r][c], Vbase + (size_t)r * Sc + kt + c);
    }
    wait_async_all();
    __syncthreads();

    // ---- S = Q K^T ----
    v8f sacc[4];
    for (int c2 = 0; c2 < 4; c2++) sacc[c2] = vz;
    for (int kk = 0; kk < HDc; kk += 32) {
      const int qrow = wv * 16 + l16;
      const v16bf af = *(const v16bf*)&sQ[qrow][kk + hw * 16];
      for (int c2 = 0; c2 < 4; c2++) {
        const v16bf bf_ = *(const v16bf*)&sK[c2 * 16 + l16][kk + hw * 16];
        sacc[c2] = WMMA_BF16(af, bf_, sacc[c2]);
      }
    }

    // ---- scale + causal mask + row max ----
    float pmax[8];
    for (int i = 0; i < 8; i++) pmax[i] = -3.0e38f;
    for (int c2 = 0; c2 < 4; c2++) {
      const int cola = kt + c2 * 16 + l16;
      for (int i = 0; i < 8; i++) {
        float svv = sacc[c2][i] * scale;
        svv = (cola <= rowa + i) ? svv : -3.0e38f;
        sacc[c2][i] = svv;
        pmax[i] = fmaxf(pmax[i], svv);
      }
    }
    for (int i = 0; i < 8; i++) {
      float v = pmax[i];
      v = fmaxf(v, __shfl_xor(v, 1, 32));
      v = fmaxf(v, __shfl_xor(v, 2, 32));
      v = fmaxf(v, __shfl_xor(v, 4, 32));
      v = fmaxf(v, __shfl_xor(v, 8, 32));
      pmax[i] = v;
    }

    // ---- online softmax ----
    float alpha[8];
    for (int i = 0; i < 8; i++) {
      const float mnew = fmaxf(m_run[i], pmax[i]);
      alpha[i] = __expf(m_run[i] - mnew);
      m_run[i] = mnew;
    }
    float rsum[8];
    for (int i = 0; i < 8; i++) rsum[i] = 0.0f;
    for (int c2 = 0; c2 < 4; c2++) {
      for (int i = 0; i < 8; i++) {
        const float p = __expf(sacc[c2][i] - m_run[i]);
        rsum[i] += p;
        sP[wv][8 * hw + i][c2 * 16 + l16] = (__bf16)p;
      }
    }
    for (int i = 0; i < 8; i++) {
      float v = rsum[i];
      v += __shfl_xor(v, 1, 32);  v += __shfl_xor(v, 2, 32);
      v += __shfl_xor(v, 4, 32);  v += __shfl_xor(v, 8, 32);
      l_run[i] = l_run[i] * alpha[i] + v;
    }
    for (int i = 0; i < 8; i++) {
      const float a = alpha[i];
      for (int j = 0; j < 8; j++) oacc[j][i] *= a;
    }

    // same-wave LDS RAW on sP: drain DScnt + compiler fence
    asm volatile("s_wait_dscnt 0x0" ::: "memory");

    // ---- O += P * V ----
    for (int ks = 0; ks < 2; ks++) {
      const v16bf af = *(const v16bf*)&sP[wv][l16][ks * 32 + hw * 16];
      for (int j = 0; j < 8; j++) {
        const v16bf bf_ = *(const v16bf*)&sVt[j * 16 + l16][ks * 32 + hw * 16];
        oacc[j] = WMMA_BF16(af, bf_, oacc[j]);
      }
    }
    __syncthreads();
  }

  // ---- normalize + write O[b][s][h*128+d] (bf16, feeds output GEMM) ----
  for (int j = 0; j < 8; j++)
    for (int i = 0; i < 8; i++) {
      const int srow = qb + wv * 16 + 8 * hw + i;
      const int col  = h * HDc + j * 16 + l16;
      Og[((size_t)b * Sc + srow) * (Hc * HDc) + col] = (__bf16)(oacc[j][i] / l_run[i]);
    }
}

// ---------------------------------------------------------------------------
// Inputs: 0=positions(i32), 1=hidden(f32), 2=w_qkv(f32), 3=w_o(f32),
//         4=q_norm_w(f32), 5=k_norm_w(f32). Output f32 [B,S,D].
// ---------------------------------------------------------------------------
extern "C" void kernel_launch(void* const* d_in, const int* in_sizes, int n_in,
                              void* d_out, int out_size, void* d_ws, size_t ws_size,
                              hipStream_t stream) {
  (void)in_sizes; (void)n_in; (void)out_size; (void)ws_size;
  const int*   positions = (const int*)d_in[0];
  const float* hidden    = (const float*)d_in[1];
  const float* w_qkv     = (const float*)d_in[2];
  const float* w_o       = (const float*)d_in[3];
  const float* qw        = (const float*)d_in[4];
  const float* kw        = (const float*)d_in[5];
  float*       out       = (float*)d_out;

  const int M  = Bc * Sc;               // 4096
  const int Nq = (Hc + 2 * KVc) * HDc;  // 6144
  const int Kd = Dc;                    // 4096

  // workspace carve-up
  char*   ws        = (char*)d_ws;
  size_t  off       = 0;
  __bf16* hidden_bf = (__bf16*)(ws + off); off += (size_t)M * Kd * 2;
  __bf16* wqkv_t    = (__bf16*)(ws + off); off += (size_t)Nq * Kd * 2;   // [Nq][Kd]
  __bf16* wo_t      = (__bf16*)(ws + off); off += (size_t)Dc * Dc * 2;   // [Dc][Dc]
  float*  qkv       = (float*) (ws + off); off += (size_t)M * Nq * 4;
  __bf16* Qb        = (__bf16*)(ws + off); off += (size_t)Bc * Hc * Sc * HDc * 2;
  __bf16* Kb        = (__bf16*)(ws + off); off += (size_t)Bc * KVc * Sc * HDc * 2;
  __bf16* Vb        = (__bf16*)(ws + off); off += (size_t)Bc * KVc * Sc * HDc * 2;
  __bf16* Obf       = (__bf16*)(ws + off);                               // [M][Dc]

  // 0) one-time conversions
  cvt_bf16_kernel<<<(M * Kd / 4 + 255) / 256, 256, 0, stream>>>(hidden, hidden_bf, M * Kd / 4);
  {
    dim3 g(Nq / 64, Kd / 64);
    cvt_transpose_bf16_kernel<<<g, 256, 0, stream>>>(w_qkv, wqkv_t, Kd, Nq);
  }
  {
    dim3 g(Dc / 64, Dc / 64);
    cvt_transpose_bf16_kernel<<<g, 256, 0, stream>>>(w_o, wo_t, Dc, Dc);
  }
  // 1) qkv = hidden @ w_qkv
  {
    dim3 grid(Nq / 128, M / 128);
    gemm_bf16_wmma<<<grid, 256, 0, stream>>>(hidden_bf, wqkv_t, qkv, M, Nq, Kd);
  }
  // 2) rmsnorm + rope -> bf16 Q/K/V (V transposed)
  {
    dim3 grid(Bc * Sc, Hc + 2 * KVc);
    norm_rope_kernel<<<grid, 128, 0, stream>>>(qkv, positions, qw, kw, Qb, Kb, Vb);
  }
  // 3) flash attention -> Obf (bf16 [B*S][H*HD])
  {
    dim3 grid(Sc / 128, Hc, Bc);
    attn_kernel<<<grid, 256, 0, stream>>>(Qb, Kb, Vb, Obf);
  }
  // 4) out = Obf @ w_o
  {
    dim3 grid(Dc / 128, M / 128);
    gemm_bf16_wmma<<<grid, 256, 0, stream>>>(Obf, wo_t, out, M, Dc, Hc * HDc);
  }
}
